// COINBlock_81415400063161
// MI455X (gfx1250) — matrix operand order repaired and checked by reading, not compiled, not run on hardware
//
#include <hip/hip_runtime.h>
#include <hip/hip_bf16.h>

// ---------------------------------------------------------------------------
// COIN block (linear-attention recurrence with per-step decay), MI455X/gfx1250.
//   Phase 1: Q/K/V = X@W + b   -> bf16 in d_ws   (WMMA f32_16x16x32_bf16)
//   Phase 2: chunkwise-parallel scan, L=32, state S[512 x 64-slice] f32 in LDS
//            K-chunk staged via Tensor Data Mover (tensor_load_to_lds) when
//            the builtin is available; plain copy fallback otherwise.
// ---------------------------------------------------------------------------

typedef __attribute__((ext_vector_type(16))) __bf16 v16bf;
typedef __attribute__((ext_vector_type(8)))  float  v8f;
typedef __attribute__((ext_vector_type(8)))  __bf16 bf16x8;
typedef __attribute__((ext_vector_type(4)))  __bf16 bf16x4;
typedef __attribute__((ext_vector_type(4)))  float  f32x4;

typedef __attribute__((ext_vector_type(4)))  unsigned int u32x4;
typedef __attribute__((ext_vector_type(8)))  int          i32x8;
typedef __attribute__((ext_vector_type(4)))  int          i32x4;

#if defined(__has_builtin)
#  if __has_builtin(__builtin_amdgcn_tensor_load_to_lds) && \
      __has_builtin(__builtin_amdgcn_s_wait_tensorcnt)
#    define USE_TDM 1
#  endif
#endif
#ifndef USE_TDM
#  define USE_TDM 0
#endif

#define DEVFN static __device__ __forceinline__

constexpr int Bb   = 4;
constexpr int Tt   = 4096;
constexpr int Cc   = 512;
constexpr int Mtot = Bb * Tt;   // 16384

DEVFN v8f vzero8() {
    v8f z;
#pragma unroll
    for (int e = 0; e < 8; ++e) z[e] = 0.0f;
    return z;
}

DEVFN v8f wmma_bf16(v16bf a, v16bf b, v8f c) {
    // D = A(16x32) * B(32x16) + C(16x16 f32)
    return __builtin_amdgcn_wmma_f32_16x16x32_bf16(
        false, a, false, b, (short)0, c, false, false);
}

// A-operand (16x32 bf16): lane L holds row M=L%16; K = {8*(L/16)+0..7, 16+8*(L/16)+0..7}
DEVFN v16bf ldA16(const __bf16* p, int lh) {
    const __bf16* q = p + 8 * lh;
    bf16x8 lo = *(const bf16x8*)(q);
    bf16x8 hi = *(const bf16x8*)(q + 16);
    v16bf r;
#pragma unroll
    for (int e = 0; e < 8; ++e) { r[e] = lo[e]; r[8 + e] = hi[e]; }
    return r;
}

// B-operand (32x16 bf16): lane L holds col N=L%16; K = {8*(L/16)+0..7, 16+8*(L/16)+0..7}
DEVFN v16bf ldB16(const __bf16* p, int lh) {
    const __bf16* q = p + 8 * lh;
    bf16x8 lo = *(const bf16x8*)(q);
    bf16x8 hi = *(const bf16x8*)(q + 16);
    v16bf r;
#pragma unroll
    for (int e = 0; e < 8; ++e) { r[e] = lo[e]; r[8 + e] = hi[e]; }
    return r;
}

// ---------------------------------------------------------------------------
// Kernel 1: Out[m,n] = bf16( X[m,:] @ W[:,n] + bias[n] ),  M=16384, N=K=512
// 256 threads = 8 waves; block tile 128x128; wave tile 32x64 (2x4 wmma tiles)
// ---------------------------------------------------------------------------
__global__ __launch_bounds__(256)
void qkv_gemm_kernel(const float* __restrict__ X,
                     const float* __restrict__ W,
                     const float* __restrict__ bias,
                     __bf16* __restrict__ Out)
{
    constexpr int KSTR = 48;                  // bf16 elems; 96B rows (16B aligned)
    __shared__ __bf16 Xs[128 * KSTR];         // [m][k]
    __shared__ __bf16 Ws[128 * KSTR];         // [n][k]  (W transposed at stage)

    const int tid  = threadIdx.x;
    const int M0   = blockIdx.x * 128;
    const int N0   = blockIdx.y * 128;
    const int w    = tid >> 5;
    const int lane = tid & 31;
    const int ln   = lane & 15;
    const int lh   = lane >> 4;
    const int Mw   = (w & 3) * 32;            // 4 M-groups of 32
    const int Nw   = (w >> 2) * 64;           // 2 N-groups of 64

    v8f acc[2][4];
#pragma unroll
    for (int mt = 0; mt < 2; ++mt)
#pragma unroll
        for (int nt = 0; nt < 4; ++nt) acc[mt][nt] = vzero8();

    for (int k0 = 0; k0 < Cc; k0 += 32) {
        // stage X tile (128 x 32), f32 -> bf16
#pragma unroll
        for (int r = 0; r < 4; ++r) {
            int gi = tid + r * 256;           // 0..1023
            int mm = gi >> 3;                 // 0..127
            int kb = (gi & 7) << 2;           // 0..28
            f32x4 xv = *(const f32x4*)(X + (size_t)(M0 + mm) * Cc + k0 + kb);
            bf16x4 o;
#pragma unroll
            for (int e = 0; e < 4; ++e) o[e] = (__bf16)xv[e];
            *(bf16x4*)(Xs + mm * KSTR + kb) = o;
        }
        // stage W tile transposed: Ws[n][kk] = W[k0+kk][N0+n]
#pragma unroll
        for (int r = 0; r < 4; ++r) {
            int gi = tid + r * 256;
            int n4 = (gi & 31) << 2;          // 0..124
            int kk = gi >> 5;                 // 0..31
            f32x4 wv = *(const f32x4*)(W + (size_t)(k0 + kk) * Cc + N0 + n4);
#pragma unroll
            for (int e = 0; e < 4; ++e)
                Ws[(n4 + e) * KSTR + kk] = (__bf16)wv[e];
        }
        __syncthreads();

        v16bf amat[2], bmat[4];
#pragma unroll
        for (int mt = 0; mt < 2; ++mt)
            amat[mt] = ldA16(Xs + (Mw + mt * 16 + ln) * KSTR, lh);
#pragma unroll
        for (int nt = 0; nt < 4; ++nt)
            bmat[nt] = ldB16(Ws + (Nw + nt * 16 + ln) * KSTR, lh);
#pragma unroll
        for (int mt = 0; mt < 2; ++mt)
#pragma unroll
            for (int nt = 0; nt < 4; ++nt)
                acc[mt][nt] = wmma_bf16(amat[mt], bmat[nt], acc[mt][nt]);
        __syncthreads();
    }

    // epilogue: + bias, cvt bf16, store. D layout: M = r + 8*lh, N = ln
#pragma unroll
    for (int nt = 0; nt < 4; ++nt) {
        const int n  = N0 + Nw + nt * 16 + ln;
        const float bn = bias[n];
#pragma unroll
        for (int mt = 0; mt < 2; ++mt) {
#pragma unroll
            for (int r = 0; r < 8; ++r) {
                int m = M0 + Mw + mt * 16 + r + 8 * lh;
                Out[(size_t)m * Cc + n] = (__bf16)(acc[mt][nt][r] + bn);
            }
        }
    }
}

// ---------------------------------------------------------------------------
// Kernel 2: chunkwise scan. Grid (C/64, B); 256 threads = 8 waves.
//   S slice [d=64][c=512] f32 resident in LDS for all 128 chunks.
// ---------------------------------------------------------------------------
__global__ __launch_bounds__(256)
void coin_scan_kernel(const __bf16* __restrict__ Qb,
                      const __bf16* __restrict__ Kb,
                      const __bf16* __restrict__ Vb,
                      float* __restrict__ Og,
                      float* __restrict__ Sout)
{
    constexpr int L = 32, DV = 64, NCHUNK = Tt / L;

    __shared__ float  Sf[DV * Cc];    // 128 KB  state slice, [d][c]
    __shared__ __bf16 Qg[L * Cc];     //  32 KB  gamma-scaled Q chunk [i][c]
    __shared__ __bf16 Ksh[L * Cc];    //  32 KB  K chunk [j][c]
    __shared__ __bf16 KT[Cc * L];     //  32 KB  (Gtot/gcum_j)-scaled K^T [c][j]
    __shared__ __bf16 VT[DV * L];     //   4 KB  V^T [d][j]
    __shared__ __bf16 As[L * L];      //   2 KB  masked intra scores [i][j]
    __shared__ float  gcum[L];
    __shared__ float  ginv[L];

    const int tid  = threadIdx.x;
    const int d0   = blockIdx.x * DV;
    const int b    = blockIdx.y;
    const int w    = tid >> 5;
    const int lane = tid & 31;
    const int ln   = lane & 15;
    const int lh   = lane >> 4;

    for (int i = tid; i < DV * Cc; i += 256) Sf[i] = 0.0f;
    __syncthreads();

    for (int cc = 0; cc < NCHUNK; ++cc) {
        const int t0 = cc * L;
        const size_t rowbase = (size_t)(b * Tt + t0) * Cc;

        // per-chunk decay cumprods: gcum[i] = prod_{j=0..i} g(t0+j)
        if (tid < L) {
            float p = 1.0f;
            for (int j = 0; j <= tid; ++j)
                p *= 0.96f + 0.03f * (float)(t0 + j) * (1.0f / 4095.0f);
            gcum[tid] = p;
            ginv[tid] = 1.0f / p;
        }
        __syncthreads();
        const float Gtot = gcum[L - 1];

        // ------------------------- stage chunk -------------------------
#if USE_TDM
        if (tid < 32) {
            // Tensor Data Mover: DMA the K chunk (32 rows x 512 bf16, row
            // stride 512) from global into LDS Ksh. D# per ISA 8.3/8.4.
            typedef __attribute__((address_space(3))) __bf16 lds_bf16_t;
            unsigned int lds_off =
                (unsigned int)(unsigned long long)(lds_bf16_t*)Ksh;
            unsigned long long ga =
                (unsigned long long)(size_t)(Kb + rowbase);
            u32x4 g0;
            g0[0] = 1u;                                    // count=1, user mode
            g0[1] = lds_off;                               // lds_addr (bytes)
            g0[2] = (unsigned int)(ga & 0xffffffffull);    // global_addr[31:0]
            g0[3] = (unsigned int)((ga >> 32) & 0x1ffffffull)
                    | (2u << 30);                          // addr[56:32] | type=2
            i32x8 g1;
            g1[0] = (int)(1u << 16);        // workgroup_mask=0, data_size=1 (2B)
            g1[1] = (int)(512u << 16);      // tensor_dim0[15:0]=512 (bits 63:48)
            g1[2] = (int)(32u << 16);       // tensor_dim0 hi=0 | tensor_dim1=32
            g1[3] = (int)(512u << 16);      // tensor_dim1 hi=0 | tile_dim0=512
            g1[4] = 32;                     // tile_dim1=32, tile_dim2=0
            g1[5] = 512;                    // tensor_dim0_stride[31:0]=512
            g1[6] = 0;                      // stride hi | tensor_dim1_stride lo
            g1[7] = 0;
            i32x4 gz = {0, 0, 0, 0};        // groups 2/3: 2D tile, dims unused
#if __clang_major__ >= 23
            i32x8 gz8 = {0, 0, 0, 0, 0, 0, 0, 0};
            __builtin_amdgcn_tensor_load_to_lds(g0, g1, gz, gz, gz8, 0);
#else
            __builtin_amdgcn_tensor_load_to_lds(g0, g1, gz, gz, 0);
#endif
        }
#else
#pragma unroll
        for (int r = 0; r < 8; ++r) {           // Ksh[j][c] straight copy
            int gi = tid + r * 256;
            int i  = gi >> 6;
            int cb = (gi & 63) << 3;
            *(bf16x8*)(Ksh + i * Cc + cb) =
                *(const bf16x8*)(Kb + rowbase + (size_t)i * Cc + cb);
        }
#endif
#pragma unroll
        for (int r = 0; r < 16; ++r) {          // Qg[i][c] = Q * gcum[i]
            int gi = tid + r * 256;
            int i  = gi >> 7;
            int cb = (gi & 127) << 2;
            bf16x4 q = *(const bf16x4*)(Qb + rowbase + (size_t)i * Cc + cb);
            float s  = gcum[i];
            bf16x4 o;
#pragma unroll
            for (int e = 0; e < 4; ++e) o[e] = (__bf16)((float)q[e] * s);
            *(bf16x4*)(Qg + i * Cc + cb) = o;
        }
#pragma unroll
        for (int r = 0; r < 16; ++r) {          // KT[c][j] = K[j][c] * Gtot/gcum[j]
            int gi = tid + r * 256;
            int j  = gi >> 7;
            int cb = (gi & 127) << 2;
            bf16x4 kv = *(const bf16x4*)(Kb + rowbase + (size_t)j * Cc + cb);
            float s   = Gtot * ginv[j];
#pragma unroll
            for (int e = 0; e < 4; ++e)
                KT[(cb + e) * L + j] = (__bf16)((float)kv[e] * s);
        }
#pragma unroll
        for (int r = 0; r < 2; ++r) {           // VT[d][j] = V[j][d0+d]
            int gi = tid + r * 256;
            int j  = gi >> 4;
            int db = (gi & 15) << 2;
            bf16x4 vv = *(const bf16x4*)(Vb + rowbase + (size_t)j * Cc + d0 + db);
#pragma unroll
            for (int e = 0; e < 4; ++e) VT[(db + e) * L + j] = vv[e];
        }
#if USE_TDM
        if (tid < 32) __builtin_amdgcn_s_wait_tensorcnt(0);
#endif
        __syncthreads();

        // ------- intra scores: A = tril( (Qg Ksh^T)[i][j] * ginv[j] ) -------
        if (w < 4) {
            const int it = w >> 1, jt = w & 1;
            v8f acc = vzero8();
            for (int kk = 0; kk < Cc; kk += 32) {
                v16bf a  = ldA16(Qg  + (it * 16 + ln) * Cc + kk, lh);
                v16bf bb = ldB16(Ksh + (jt * 16 + ln) * Cc + kk, lh);
                acc = wmma_bf16(a, bb, acc);
            }
            const int j    = jt * 16 + ln;
            const float iv = ginv[j];
#pragma unroll
            for (int r = 0; r < 8; ++r) {
                int i   = it * 16 + r + 8 * lh;
                float v = (j <= i) ? acc[r] * iv : 0.0f;
                As[i * L + j] = (__bf16)v;
            }
        } else if (cc + 1 < NCHUNK) {
            // idle waves: warm L2 with next chunk's Q/K rows
            const size_t nb  = rowbase + (size_t)L * Cc;
            const int    row = (w - 4) * 8 + (lane >> 2);
            const size_t off = nb + (size_t)row * Cc + (size_t)(lane & 3) * 128;
            __builtin_prefetch(Qb + off, 0, 1);
            __builtin_prefetch(Kb + off, 0, 1);
        }
        __syncthreads();

        // --------- O tile: cross (Qg @ S_in) + intra (A @ V), tanh ---------
        {
            const int it = w >> 2, dt = w & 3;
            const int i  = it * 16 + ln;
            const int d  = dt * 16 + ln;
            v8f acc = vzero8();
            for (int kk = 0; kk < Cc; kk += 32) {
                v16bf a = ldA16(Qg + i * Cc + kk, lh);
                // B-operand from f32 state: c = kk + {8*lh + 0..7, 16 + 8*lh + 0..7}
                const float* sp = Sf + d * Cc + kk + 8 * lh;
                f32x4 s0 = *(const f32x4*)(sp);
                f32x4 s1 = *(const f32x4*)(sp + 4);
                f32x4 s2 = *(const f32x4*)(sp + 16);
                f32x4 s3 = *(const f32x4*)(sp + 20);
                v16bf bb;
#pragma unroll
                for (int e = 0; e < 4; ++e) {
                    bb[e]      = (__bf16)s0[e];
                    bb[4 + e]  = (__bf16)s1[e];
                    bb[8 + e]  = (__bf16)s2[e];
                    bb[12 + e] = (__bf16)s3[e];
                }
                acc = wmma_bf16(a, bb, acc);
            }
            // intra: single k=32 step over j
            v16bf a2 = ldA16(As + i * L, lh);
            v16bf b2 = ldB16(VT + d * L, lh);
            acc = wmma_bf16(a2, b2, acc);
#pragma unroll
            for (int r = 0; r < 8; ++r) {
                int t = t0 + it * 16 + r + 8 * lh;
                Og[(size_t)(b * Tt + t) * Cc + d0 + d] = tanhf(acc[r]);
            }
        }
        __syncthreads();   // all reads of S_in done before update writes

        // --------- state update: S[c][d] = Gtot*S + (K'^T V)[c][d] ---------
        for (int q = 0; q < 16; ++q) {
            int tt = w * 16 + q;              // 0..127 tiles (32 c-tiles x 4 d-tiles)
            int ct = tt >> 2, dt = tt & 3;
            v16bf a  = ldA16(KT + (ct * 16 + ln) * L, lh);
            v16bf bb = ldB16(VT + (dt * 16 + ln) * L, lh);
            v8f up = wmma_bf16(a, bb, vzero8());
            const int d_ = dt * 16 + ln;
#pragma unroll
            for (int r = 0; r < 8; ++r) {
                int c_ = ct * 16 + r + 8 * lh;
                float* sp = &Sf[d_ * Cc + c_];
                *sp = up[r] + Gtot * (*sp);
            }
        }
        __syncthreads();
    }

    // write final state: Sout[b][c][d0+d]
    for (int idx = tid; idx < DV * Cc; idx += 256) {
        int d = idx >> 9;
        int c = idx & 511;
        Sout[(size_t)b * Cc * Cc + (size_t)c * Cc + d0 + d] = Sf[d * Cc + c];
    }
}

// ---------------------------------------------------------------------------
extern "C" void kernel_launch(void* const* d_in, const int* in_sizes, int n_in,
                              void* d_out, int out_size, void* d_ws, size_t ws_size,
                              hipStream_t stream) {
    (void)in_sizes; (void)n_in; (void)out_size; (void)ws_size;

    const float* X  = (const float*)d_in[0];
    // d_in[1] = S_n: reference resets it to zeros -> unused
    const float* WQ = (const float*)d_in[2];
    const float* bQ = (const float*)d_in[3];
    const float* WK = (const float*)d_in[4];
    const float* bK = (const float*)d_in[5];
    const float* WV = (const float*)d_in[6];
    const float* bV = (const float*)d_in[7];

    // bf16 Q/K/V intermediates in workspace: 3 * 16384*512 * 2B = 48 MB
    __bf16* Qb = (__bf16*)d_ws;
    __bf16* Kb = Qb + (size_t)Mtot * Cc;
    __bf16* Vb = Kb + (size_t)Mtot * Cc;

    float* Og   = (float*)d_out;
    float* Sout = Og + (size_t)Bb * Tt * Cc;

    dim3 g1(Mtot / 128, Cc / 128, 1);
    qkv_gemm_kernel<<<g1, 256, 0, stream>>>(X, WQ, bQ, Qb);
    qkv_gemm_kernel<<<g1, 256, 0, stream>>>(X, WK, bK, Kb);
    qkv_gemm_kernel<<<g1, 256, 0, stream>>>(X, WV, bV, Vb);

    dim3 g2(Cc / 64, Bb, 1);
    coin_scan_kernel<<<g2, 256, 0, stream>>>(Qb, Kb, Vb, Og, Sout);
}
